// KBLaMBitNetAttention_52982716563902
// MI455X (gfx1250) — matrix operand order repaired and checked by reading, not compiled, not run on hardware
//
#include <hip/hip_runtime.h>

// ---------------------------------------------------------------------------
// KBLaM rectangular attention for MI455X (gfx1250, wave32, WMMA bf16->f32).
// Pipeline: f32->bf16 convert, WMMA GEMM projections (async-LDS staged B),
// RoPE/transpose packs, flash-style joint-softmax attention (KB + causal
// self), WMMA out-proj.
// ---------------------------------------------------------------------------

typedef __bf16 bf16_t;
typedef __attribute__((ext_vector_type(16))) __bf16 v16bf;
typedef __attribute__((ext_vector_type(8)))  __bf16 v8bf;
typedef __attribute__((ext_vector_type(8)))  float  v8f;

#define BATCH 2
#define SEQ   1024
#define HIDN  2048
#define NHEAD 32
#define KVHN  8
#define HD    64
#define KBN   512
#define KBL   513   // KB keys + sink
#define KBLP  544   // padded key stride for V^T loads (multiple of 32)

#define WMMA_BF16(a, b, c) \
  __builtin_amdgcn_wmma_f32_16x16x32_bf16(false, (a), false, (b), (short)0, (c), false, false)

// 16 contiguous bf16 -> one A/B fragment register set (two b128 loads).
__device__ __forceinline__ v16bf ld_contig16(const bf16_t* p) {
  v8bf lo = *(const v8bf*)(p);
  v8bf hi = *(const v8bf*)(p + 8);
  v16bf r;
#pragma unroll
  for (int i = 0; i < 8; ++i) { r[i] = lo[i]; r[i + 8] = hi[i]; }
  return r;
}
// Two disjoint 8-element chunks (A-matrix layout: K = hh*8+i and 16+hh*8+i).
__device__ __forceinline__ v16bf ld_split8(const bf16_t* p0, const bf16_t* p1) {
  v8bf lo = *(const v8bf*)(p0);
  v8bf hi = *(const v8bf*)(p1);
  v16bf r;
#pragma unroll
  for (int i = 0; i < 8; ++i) { r[i] = lo[i]; r[i + 8] = hi[i]; }
  return r;
}

// CDNA5 async copy: each lane moves 16B global -> LDS, tracked by ASYNCcnt.
__device__ __forceinline__ void async_load_b128(unsigned lds_off, const void* gptr) {
  asm volatile("global_load_async_to_lds_b128 %0, %1, off"
               :: "v"(lds_off), "v"(gptr)
               : "memory");
}
__device__ __forceinline__ void wait_async0() {
  asm volatile("s_wait_asynccnt 0" ::: "memory");
}

// ---------------------------------------------------------------------------
// fp32 -> bf16 conversion
// ---------------------------------------------------------------------------
__global__ void cvt_f32_bf16(const float* __restrict__ in, bf16_t* __restrict__ out, int n) {
  for (int i = blockIdx.x * blockDim.x + threadIdx.x; i < n; i += gridDim.x * blockDim.x)
    out[i] = (bf16_t)in[i];
}

// ---------------------------------------------------------------------------
// GEMM: C[M,N] = A[M,K] @ W[N,K]^T.  A,W bf16 row-major, f32 accumulate.
// Block = 256 threads = 8 waves. Wave: 32(M) x 64(N). WG: 256(M) x 64(N).
// The 64x32 B tile (4KB) is shared by all 8 waves: staged in LDS with
// double-buffered global_load_async_to_lds_b128 (one b128 per thread).
// grid = (N/64, M/256). K multiple of 32.
// ---------------------------------------------------------------------------
template <bool F32OUT>
__global__ __launch_bounds__(256) void gemm_bf16(const bf16_t* __restrict__ A,
                                                 const bf16_t* __restrict__ W,
                                                 bf16_t* __restrict__ Cb,
                                                 float* __restrict__ Cf,
                                                 int M, int N, int K) {
  __shared__ __align__(16) bf16_t bsh[2][64][32];
  const int lane = threadIdx.x & 31;
  const int w    = threadIdx.x >> 5;
  const int mBase = blockIdx.y * 256 + w * 32;
  const int nBase = blockIdx.x * 64;
  const int lo = lane & 15;   // A: row-in-tile / B: col-in-tile / C: N
  const int hh = lane >> 4;   // A/B: K-half select   / C: M-half select

  // async staging assignment: thread -> (N row, 8-elem K segment) of B tile
  const int sn = threadIdx.x >> 2;        // 0..63
  const int sk = (threadIdx.x & 3) * 8;   // 0,8,16,24
  const bf16_t* wrow = W + (size_t)(nBase + sn) * K + sk;
  const unsigned lds0 = (unsigned)(unsigned long long)(const void*)&bsh[0][sn][sk];
  const unsigned lds1 = (unsigned)(unsigned long long)(const void*)&bsh[1][sn][sk];

  v8f acc[2][4];
#pragma unroll
  for (int t = 0; t < 2; ++t)
#pragma unroll
    for (int j = 0; j < 4; ++j)
#pragma unroll
      for (int r = 0; r < 8; ++r) acc[t][j][r] = 0.0f;

  const bf16_t* arow0 = A + (size_t)(mBase + lo) * K;
  const bf16_t* arow1 = A + (size_t)(mBase + 16 + lo) * K;

  async_load_b128(lds0, wrow);   // prime buffer 0 with kk=0 tile
  wait_async0();
  __syncthreads();

  for (int kk = 0; kk < K; kk += 32) {
    const int cur = (kk >> 5) & 1;
    if (kk + 32 < K)  // overlap next B tile copy with this step's WMMAs
      async_load_b128(cur ? lds0 : lds1, wrow + kk + 32);
    // A fragments: lane (m + 16*hh) holds K = kk+hh*16+i
    v16bf af0 = ld_contig16(arow0 + kk + hh * 16);
    v16bf af1 = ld_contig16(arow1 + kk + hh * 16);
    __builtin_prefetch(arow0 + kk + 512, 0, 1);  // global_prefetch_b8
#pragma unroll
    for (int j = 0; j < 4; ++j) {
      // B fragment from LDS: lane (n + 16*h2) reads bsh[cur][j*16+n][h2*16..+15]
      v16bf bf = ld_contig16(&bsh[cur][j * 16 + lo][hh * 16]);
      acc[0][j] = WMMA_BF16(af0, bf, acc[0][j]);
      acc[1][j] = WMMA_BF16(af1, bf, acc[1][j]);
    }
    wait_async0();     // next tile landed (and our own copy done)
    __syncthreads();   // all waves done reading cur before it is reused
  }
  // C layout: lane L -> M = (L>>4)*8 + r, N = L&15
#pragma unroll
  for (int t = 0; t < 2; ++t)
#pragma unroll
    for (int j = 0; j < 4; ++j)
#pragma unroll
      for (int r = 0; r < 8; ++r) {
        const int row = mBase + t * 16 + hh * 8 + r;
        const int col = nBase + j * 16 + lo;
        if (F32OUT) Cf[(size_t)row * N + col] = acc[t][j][r];
        else        Cb[(size_t)row * N + col] = (bf16_t)acc[t][j][r];
      }
}

// ---------------------------------------------------------------------------
// RoPE + pack [B*L, H*HD] -> [B, H, outLen, HD] (key-major)
// ---------------------------------------------------------------------------
__global__ void rope_pack(const bf16_t* __restrict__ in, bf16_t* __restrict__ out,
                          const float* __restrict__ cosT, const float* __restrict__ sinT,
                          int H, int L, int outLen, int keyOff, int useRope) {
  const int total = BATCH * L * H * HD;
  int i = blockIdx.x * blockDim.x + threadIdx.x;
  if (i >= total) return;
  const int d   = i & 63;
  const int h   = (i >> 6) % H;
  const int pos = ((i >> 6) / H) % L;
  const int b   = i / (HD * H * L);
  float v = (float)in[i];
  float o = v;
  if (useRope) {
    float prt = (float)in[(d < 32) ? (i + 32) : (i - 32)];
    float c = cosT[pos * HD + d];
    float s = sinT[pos * HD + d];
    o = (d < 32) ? (v * c - prt * s) : (v * c + prt * s);
  }
  out[((size_t)(b * H + h) * outLen + pos + keyOff) * HD + d] = (bf16_t)o;
}

// pack [B*L, H*HD] -> [B, H, HD, strideS] (d-major, for PV B-fragments)
__global__ void packT(const bf16_t* __restrict__ in, bf16_t* __restrict__ out,
                      int H, int L, int strideS, int keyOff) {
  const int total = BATCH * L * H * HD;
  int i = blockIdx.x * blockDim.x + threadIdx.x;
  if (i >= total) return;
  const int d   = i & 63;
  const int h   = (i >> 6) % H;
  const int pos = ((i >> 6) / H) % L;
  const int b   = i / (HD * H * L);
  out[((size_t)(b * H + h) * HD + d) * strideS + pos + keyOff] = (bf16_t)in[i];
}

// Broadcast sink key/value into slot 0 of kbk_t / kbvT.
__global__ void sink_pack(const float* __restrict__ sk, const float* __restrict__ sv,
                          bf16_t* __restrict__ kbk_t, bf16_t* __restrict__ kbvT) {
  int i = blockIdx.x * blockDim.x + threadIdx.x;
  if (i >= BATCH * KVHN * HD) return;
  const int d = i & 63;
  const int h = (i >> 6) & 7;
  const int b = i >> 9;
  kbk_t[(size_t)((b * KVHN + h) * KBL) * HD + d]     = (bf16_t)sk[h * HD + d];
  kbvT[((size_t)(b * KVHN + h) * HD + d) * KBLP + 0] = (bf16_t)sv[h * HD + d];
}

// ---------------------------------------------------------------------------
// Flash-style attention with joint softmax over [KB(513) | causal self].
// One wave per (b, head, 16-row q tile). 32-key chunks:
//   scores: 2 n-tiles x 2 d-steps of WMMA; online softmax with half-wave
//   shuffles; P transposed C->A layout through LDS (s_wait_dscnt); 4 PV WMMAs.
// ---------------------------------------------------------------------------
__global__ __launch_bounds__(256) void kblam_attn_kernel(
    const bf16_t* __restrict__ q_t, const bf16_t* __restrict__ kbq_t,
    const bf16_t* __restrict__ k_t, const bf16_t* __restrict__ vT,
    const bf16_t* __restrict__ kbk_t, const bf16_t* __restrict__ kbvT,
    bf16_t* __restrict__ attn_out) {
  __shared__ __align__(16) bf16_t pbuf[8][16][32];  // per-wave P tile
  const int w    = threadIdx.x >> 5;
  const int lane = threadIdx.x & 31;
  const int task = blockIdx.x * 8 + w;            // 4096 tasks
  const int b  = task >> 11;
  const int h  = (task >> 6) & 31;
  const int q0 = (task & 63) << 4;
  const int kvh = h >> 2;                         // GQA: NH/KVH = 4
  const int lo = lane & 15;
  const int hh = lane >> 4;

  bf16_t (*pb)[32] = pbuf[w];

  // A fragments for Q (self) and KBQ (knowledge base), d split 0..31 / 32..63
  const bf16_t* qrow  = q_t   + ((size_t)(b * NHEAD + h) * SEQ + q0 + lo) * HD;
  const bf16_t* qbrow = kbq_t + ((size_t)(b * NHEAD + h) * SEQ + q0 + lo) * HD;
  v16bf aq0 = ld_contig16(qrow + hh * 16);
  v16bf aq1 = ld_contig16(qrow + 32 + hh * 16);
  v16bf ab0 = ld_contig16(qbrow + hh * 16);
  v16bf ab1 = ld_contig16(qbrow + 32 + hh * 16);

  v8f o[4];
#pragma unroll
  for (int dt = 0; dt < 4; ++dt)
#pragma unroll
    for (int r = 0; r < 8; ++r) o[dt][r] = 0.0f;
  float mrow[8], lrow[8];
#pragma unroll
  for (int r = 0; r < 8; ++r) { mrow[r] = -3.0e38f; lrow[r] = 0.0f; }

  auto chunk = [&](v16bf a0, v16bf a1, const bf16_t* kb, const bf16_t* vb,
                   int vstride, int kc, int klimit, bool causal) {
    v8f s[2];
#pragma unroll
    for (int nt = 0; nt < 2; ++nt) {
      int key  = kc + nt * 16 + lo;
      int krow = (key > klimit) ? klimit : key;   // clamp OOB rows (masked below)
      const bf16_t* kp = kb + (size_t)krow * HD + hh * 16;
      v16bf b0 = ld_contig16(kp);        // d 0..31 half (per lane hh)
      v16bf b1 = ld_contig16(kp + 32);   // d 32..63 half
      v8f c;
#pragma unroll
      for (int r = 0; r < 8; ++r) c[r] = 0.0f;
      c = WMMA_BF16(a0, b0, c);
      c = WMMA_BF16(a1, b1, c);
      s[nt] = c;
    }
    // scale + mask, per-row chunk max
    float cmax[8];
#pragma unroll
    for (int r = 0; r < 8; ++r) {
      const int qpos = q0 + hh * 8 + r;
#pragma unroll
      for (int nt = 0; nt < 2; ++nt) {
        const int key = kc + nt * 16 + lo;
        float sv = s[nt][r] * 0.125f;   // 1/sqrt(64)
        const bool ok = causal ? (key <= qpos) : (key <= klimit);
        s[nt][r] = ok ? sv : -1.0e30f;
      }
      cmax[r] = fmaxf(s[0][r], s[1][r]);
    }
#pragma unroll
    for (int offs = 1; offs < 16; offs <<= 1)
#pragma unroll
      for (int r = 0; r < 8; ++r)
        cmax[r] = fmaxf(cmax[r], __shfl_xor(cmax[r], offs, 32));
    // online softmax update
    float alpha[8], psum[8];
#pragma unroll
    for (int r = 0; r < 8; ++r) {
      const float mn = fmaxf(mrow[r], cmax[r]);
      alpha[r] = __expf(mrow[r] - mn);
      mrow[r] = mn;
    }
#pragma unroll
    for (int r = 0; r < 8; ++r) {
      const float p0 = __expf(s[0][r] - mrow[r]);
      const float p1 = __expf(s[1][r] - mrow[r]);
      s[0][r] = p0; s[1][r] = p1;
      psum[r] = p0 + p1;
    }
#pragma unroll
    for (int offs = 1; offs < 16; offs <<= 1)
#pragma unroll
      for (int r = 0; r < 8; ++r)
        psum[r] += __shfl_xor(psum[r], offs, 32);
#pragma unroll
    for (int r = 0; r < 8; ++r) lrow[r] = lrow[r] * alpha[r] + psum[r];
#pragma unroll
    for (int dt = 0; dt < 4; ++dt)
#pragma unroll
      for (int r = 0; r < 8; ++r) o[dt][r] *= alpha[r];
    // P: C layout -> LDS (rows x 32 keys) -> A layout
#pragma unroll
    for (int nt = 0; nt < 2; ++nt)
#pragma unroll
      for (int r = 0; r < 8; ++r)
        pb[hh * 8 + r][nt * 16 + lo] = (bf16_t)s[nt][r];
    asm volatile("s_wait_dscnt 0" ::: "memory");
    v16bf pf = ld_contig16(&pb[lo][hh * 16]);
    // P @ V: B fragment is d-major V^T (contiguous lane loads)
#pragma unroll
    for (int dt = 0; dt < 4; ++dt) {
      const bf16_t* vp = vb + (size_t)(dt * 16 + lo) * vstride + kc + hh * 16;
      v16bf vf = ld_contig16(vp);
      o[dt] = WMMA_BF16(pf, vf, o[dt]);
    }
  };

  // Phase 1: KB keys (sink at 0 + 512 entries = 513), 17 chunks of 32
  const bf16_t* kbK = kbk_t + (size_t)(b * KVHN + kvh) * KBL * HD;
  const bf16_t* kbV = kbvT  + (size_t)(b * KVHN + kvh) * HD * KBLP;
  for (int c = 0; c < 17; ++c) chunk(ab0, ab1, kbK, kbV, KBLP, c * 32, KBL - 1, false);

  // Phase 2: causal self keys up to q0+15
  const bf16_t* sK = k_t + (size_t)(b * KVHN + kvh) * SEQ * HD;
  const bf16_t* sV = vT  + (size_t)(b * KVHN + kvh) * HD * SEQ;
  const int nch = (q0 + 47) >> 5;
  for (int c = 0; c < nch; ++c) chunk(aq0, aq1, sK, sV, SEQ, c * 32, SEQ - 1, true);

  // Normalize and store [B*S, NH*HD]
#pragma unroll
  for (int dt = 0; dt < 4; ++dt)
#pragma unroll
    for (int r = 0; r < 8; ++r) {
      const float val = o[dt][r] / lrow[r];
      const int row = b * SEQ + q0 + hh * 8 + r;
      const int col = h * HD + dt * 16 + lo;
      attn_out[(size_t)row * (NHEAD * HD) + col] = (bf16_t)val;
    }
}

// ---------------------------------------------------------------------------
// Host-side launch
// ---------------------------------------------------------------------------
extern "C" void kernel_launch(void* const* d_in, const int* in_sizes, int n_in,
                              void* d_out, int out_size, void* d_ws, size_t ws_size,
                              hipStream_t stream) {
  (void)in_sizes; (void)n_in; (void)out_size; (void)ws_size;
  const float* hidden  = (const float*)d_in[0];
  const float* cosT    = (const float*)d_in[1];
  const float* sinT    = (const float*)d_in[2];
  const float* kb_keys = (const float*)d_in[3];
  const float* kb_vals = (const float*)d_in[4];
  const float* Wq      = (const float*)d_in[5];
  const float* Wk      = (const float*)d_in[6];
  const float* Wv      = (const float*)d_in[7];
  const float* Wqn     = (const float*)d_in[8];
  const float* Wkbk    = (const float*)d_in[9];
  const float* Wkbv    = (const float*)d_in[10];
  const float* Wo      = (const float*)d_in[11];
  const float* sinkK   = (const float*)d_in[12];
  const float* sinkV   = (const float*)d_in[13];
  float* out = (float*)d_out;

  char* ws = (char*)d_ws;
  size_t off = 0;
  auto alloc = [&](size_t elems) {
    bf16_t* p = (bf16_t*)(ws + off);
    off += (elems * sizeof(bf16_t) + 255) & ~(size_t)255;
    return p;
  };
  const size_t T  = (size_t)BATCH * SEQ;   // 2048 tokens
  const size_t TK = (size_t)BATCH * KBN;   // 1024 kb rows

  bf16_t* h_bf    = alloc(T * HIDN);
  bf16_t* wq_bf   = alloc((size_t)HIDN * HIDN);
  bf16_t* wk_bf   = alloc((size_t)KVHN * HD * HIDN);
  bf16_t* wv_bf   = alloc((size_t)KVHN * HD * HIDN);
  bf16_t* wqn_bf  = alloc((size_t)HIDN * HIDN);
  bf16_t* wkbk_bf = alloc((size_t)KVHN * HD * HIDN);
  bf16_t* wkbv_bf = alloc((size_t)KVHN * HD * HIDN);
  bf16_t* wo_bf   = alloc((size_t)HIDN * HIDN);
  bf16_t* kbkin   = alloc(TK * HIDN);
  bf16_t* kbvin   = alloc(TK * HIDN);
  bf16_t* proj_q  = alloc(T * HIDN);
  bf16_t* proj_qn = alloc(T * HIDN);
  bf16_t* proj_k  = alloc(T * KVHN * HD);
  bf16_t* proj_v  = alloc(T * KVHN * HD);
  bf16_t* proj_kk = alloc(TK * KVHN * HD);
  bf16_t* proj_kv = alloc(TK * KVHN * HD);
  bf16_t* q_t     = alloc((size_t)BATCH * NHEAD * SEQ * HD);
  bf16_t* kbq_t   = alloc((size_t)BATCH * NHEAD * SEQ * HD);
  bf16_t* k_t     = alloc((size_t)BATCH * KVHN * SEQ * HD);
  bf16_t* kbk_t   = alloc((size_t)BATCH * KVHN * KBL * HD);
  bf16_t* vT      = alloc((size_t)BATCH * KVHN * HD * SEQ);
  bf16_t* kbvT    = alloc((size_t)BATCH * KVHN * HD * KBLP);
  bf16_t* attn_o  = alloc(T * NHEAD * HD);

  auto cvt = [&](const float* src, bf16_t* dst, size_t n) {
    int blocks = (int)((n + 255) / 256); if (blocks > 8192) blocks = 8192;
    cvt_f32_bf16<<<blocks, 256, 0, stream>>>(src, dst, (int)n);
  };
  cvt(hidden,  h_bf,    T * HIDN);
  cvt(Wq,      wq_bf,   (size_t)HIDN * HIDN);
  cvt(Wk,      wk_bf,   (size_t)KVHN * HD * HIDN);
  cvt(Wv,      wv_bf,   (size_t)KVHN * HD * HIDN);
  cvt(Wqn,     wqn_bf,  (size_t)HIDN * HIDN);
  cvt(Wkbk,    wkbk_bf, (size_t)KVHN * HD * HIDN);
  cvt(Wkbv,    wkbv_bf, (size_t)KVHN * HD * HIDN);
  cvt(Wo,      wo_bf,   (size_t)HIDN * HIDN);
  cvt(kb_keys, kbkin,   TK * HIDN);
  cvt(kb_vals, kbvin,   TK * HIDN);

  auto gemmB = [&](const bf16_t* A, const bf16_t* W, bf16_t* C, int M, int N, int K) {
    dim3 g(N / 64, M / 256);
    gemm_bf16<false><<<g, 256, 0, stream>>>(A, W, C, nullptr, M, N, K);
  };
  gemmB(h_bf,  wq_bf,   proj_q,  (int)T,  HIDN,      HIDN);
  gemmB(h_bf,  wqn_bf,  proj_qn, (int)T,  HIDN,      HIDN);
  gemmB(h_bf,  wk_bf,   proj_k,  (int)T,  KVHN * HD, HIDN);
  gemmB(h_bf,  wv_bf,   proj_v,  (int)T,  KVHN * HD, HIDN);
  gemmB(kbkin, wkbk_bf, proj_kk, (int)TK, KVHN * HD, HIDN);
  gemmB(kbvin, wkbv_bf, proj_kv, (int)TK, KVHN * HD, HIDN);

  // RoPE + packs
  rope_pack<<<(BATCH * SEQ * NHEAD * HD) / 256, 256, 0, stream>>>(
      proj_q, q_t, cosT, sinT, NHEAD, SEQ, SEQ, 0, 1);
  rope_pack<<<(BATCH * SEQ * NHEAD * HD) / 256, 256, 0, stream>>>(
      proj_qn, kbq_t, cosT, sinT, NHEAD, SEQ, SEQ, 0, 0);
  rope_pack<<<(BATCH * SEQ * KVHN * HD) / 256, 256, 0, stream>>>(
      proj_k, k_t, cosT, sinT, KVHN, SEQ, SEQ, 0, 1);
  rope_pack<<<(BATCH * KBN * KVHN * HD) / 256, 256, 0, stream>>>(
      proj_kk, kbk_t, cosT, sinT, KVHN, KBN, KBL, 1, 0);
  packT<<<(BATCH * SEQ * KVHN * HD) / 256, 256, 0, stream>>>(
      proj_v, vT, KVHN, SEQ, SEQ, 0);
  packT<<<(BATCH * KBN * KVHN * HD) / 256, 256, 0, stream>>>(
      proj_kv, kbvT, KVHN, KBN, KBLP, 1);
  sink_pack<<<4, 256, 0, stream>>>(sinkK, sinkV, kbk_t, kbvT);

  // Attention: 4096 wave-tasks, 8 waves per block
  kblam_attn_kernel<<<512, 256, 0, stream>>>(q_t, kbq_t, k_t, vT, kbk_t, kbvT, attn_o);

  // Output projection straight to f32 d_out
  dim3 gf(HIDN / 64, (int)T / 256);
  gemm_bf16<true><<<gf, 256, 0, stream>>>(attn_o, wo_bf, nullptr, out, (int)T, HIDN, HIDN);
}
